// Dynamic_conv2d_17154099380476
// MI455X (gfx1250) — matrix operand order repaired and checked by reading, not compiled, not run on hardware
//
#include <hip/hip_runtime.h>

typedef __attribute__((ext_vector_type(16))) __bf16 v16bf;
typedef __attribute__((ext_vector_type(8)))  float  v8f;

#define B_   32
#define D_   512
#define HID_ 256
#define NF_  8
#define CIN_ 64
#define COUT_ 64
#define HW_  64
#define KK_  576              // CIN*3*3
#define WPS_ (COUT_ * KK_)    // 36864 elements per filter / per-sample agg matrix
#define NTILE_ 128            // pixels per block (2 image rows)

// ---------------- Kernel 1: routing MLP + softmax ----------------
__global__ __launch_bounds__(256) void routing_kernel(
    const float* __restrict__ x,  const float* __restrict__ w1,
    const float* __restrict__ b1, const float* __restrict__ w2,
    const float* __restrict__ b2, float* __restrict__ attn_ws,
    float* __restrict__ attn_out) {
  __shared__ float h[HID_];
  __shared__ float logits[NF_];
  const int b = blockIdx.x;
  const int t = threadIdx.x;
  const float* xb  = x  + b * D_;
  const float* w1r = w1 + t * D_;
  float acc = b1[t];
  for (int i = 0; i < D_; ++i) acc = fmaf(xb[i], w1r[i], acc);
  h[t] = fmaxf(acc, 0.0f);
  __syncthreads();
  if (t < NF_) {
    const float* w2r = w2 + t * HID_;
    float a = b2[t];
    for (int i = 0; i < HID_; ++i) a = fmaf(h[i], w2r[i], a);
    logits[t] = a * (1.0f / 30.0f);   // temperature
  }
  __syncthreads();
  if (t == 0) {
    float m = logits[0];
    for (int f = 1; f < NF_; ++f) m = fmaxf(m, logits[f]);
    float e[NF_]; float s = 0.0f;
    for (int f = 0; f < NF_; ++f) { e[f] = __expf(logits[f] - m); s += e[f]; }
    const float inv = 1.0f / s;
    for (int f = 0; f < NF_; ++f) {
      const float v = e[f] * inv;
      attn_ws[b * NF_ + f]  = v;
      attn_out[b * NF_ + f] = v;   // tuple output #2
    }
  }
}

// ------- Kernel 2: aggregate filter bank -> per-sample bf16 GEMM A -------
__global__ __launch_bounds__(256) void agg_kernel(
    const float* __restrict__ attn, const float* __restrict__ weight,
    __bf16* __restrict__ aggw) {
  const long idx = (long)blockIdx.x * 256 + threadIdx.x;  // (b, j)
  const long total = (long)B_ * WPS_;
  if (idx >= total) return;
  const int b = (int)(idx / WPS_);
  const int j = (int)(idx - (long)b * WPS_);
  const float* at = attn + b * NF_;
  float a = 0.0f;
#pragma unroll
  for (int f = 0; f < NF_; ++f) a = fmaf(at[f], weight[(long)f * WPS_ + j], a);
  aggw[idx] = (__bf16)a;
}

// ------- Kernel 3: implicit-GEMM grouped conv via bf16 WMMA -------
// Block = (sample b, pair of image rows). Tile: M=64 (Cout) x N=128 px, K=576 in 18x32.
// 8 waves, each a 32x32 register tile => 4 v_wmma per K-chunk with 2x A/B fragment reuse.
__global__ __launch_bounds__(256) void conv_wmma_kernel(
    const float* __restrict__ xin, const __bf16* __restrict__ aggw,
    const float* __restrict__ attn, const float* __restrict__ bias_p,
    float* __restrict__ out) {
  __shared__ __align__(16) __bf16 As[COUT_][32];    // [M][k_local]
  __shared__ __align__(16) __bf16 Bs[NTILE_][32];   // [pixel][k_local] (transposed im2col)

  const int blk  = blockIdx.x;
  const int b    = blk >> 5;        // / 32 row-pairs
  const int y0   = (blk & 31) * 2;  // first of two image rows
  const int t    = threadIdx.x;
  const int lane = t & 31;
  const int wv   = t >> 5;          // 8 waves: 2 along M x 4 along N
  const int mo   = (wv & 1) * 32;
  const int no   = (wv >> 1) * 32;

  const __bf16* Abase = aggw + (long)b * WPS_;
  const float*  Xb    = xin  + (long)b * CIN_ * HW_ * HW_;

  // A loader: thread -> (row, 8-half segment)
  const int arow = t >> 2;              // 0..63
  const int aseg = (t & 3) * 8;         // 0,8,16,24
  // B loader: thread -> (k_local, 16 consecutive pixels)
  const int kl = t >> 3;                // 0..31
  const int pg = (t & 7) * 16;          // pixel start: 0..112

  v8f acc00 = {}; v8f acc01 = {}; v8f acc10 = {}; v8f acc11 = {};

  for (int kc = 0; kc < 18; ++kc) {
    const int kbase = kc * 32;
    // --- stage A chunk (vector 16B copies) ---
    *(uint4*)(&As[arow][aseg]) =
        *(const uint4*)(Abase + (long)arow * KK_ + kbase + aseg);
    // --- stage B chunk: im2col with zero padding, two image rows ---
    {
      const int k   = kbase + kl;
      const int cin = k / 9;
      const int rem = k - cin * 9;
      const int kh  = rem / 3;
      const int kw  = rem - kh * 3;
#pragma unroll
      for (int i = 0; i < 16; ++i) {
        const int p  = pg + i;
        const int y  = y0 + (p >> 6);
        const int x  = p & 63;
        const int ry = y + kh - 1;
        const int rx = x + kw - 1;
        const bool ok = (ry >= 0) & (ry < HW_) & (rx >= 0) & (rx < HW_);
        const float v = ok ? Xb[((long)cin * HW_ + ry) * HW_ + rx] : 0.0f;
        Bs[p][kl] = (__bf16)v;
      }
    }
    // --- prefetch next K-chunk's global data (global_prefetch_b8 hint) ---
    if (kc + 1 < 18) {
      __builtin_prefetch(Abase + (long)arow * KK_ + kbase + 32 + aseg, 0, 1);
      const int kn   = kbase + 32 + kl;
      const int cinn = kn / 9;
      __builtin_prefetch(Xb + ((long)cinn * HW_ + y0) * HW_ + (pg & 63), 0, 1);
    }
    __syncthreads();

    // --- assemble fragments, contiguous k = 16h + j in A and B ---
    v16bf a0, a1;
    {
      const int r   = lane & 15;
      const int kb16 = (lane >= 16) ? 16 : 0;
#pragma unroll
      for (int j = 0; j < 16; ++j) {
        a0[j] = As[mo + r][kb16 + j];               // K 0..15 / 16..31
        a1[j] = As[mo + 16 + r][kb16 + j];
      }
    }
    v16bf b0, b1;
    {
      const int n0   = no + (lane & 15);
      const int kb16 = (lane >= 16) ? 16 : 0;
#pragma unroll
      for (int j = 0; j < 16; ++j) {
        b0[j] = Bs[n0][kb16 + j];
        b1[j] = Bs[n0 + 16][kb16 + j];
      }
    }
    acc00 = __builtin_amdgcn_wmma_f32_16x16x32_bf16(false, a0, false, b0,
                                                    (short)0, acc00, false, false);
    acc01 = __builtin_amdgcn_wmma_f32_16x16x32_bf16(false, a0, false, b1,
                                                    (short)0, acc01, false, false);
    acc10 = __builtin_amdgcn_wmma_f32_16x16x32_bf16(false, a1, false, b0,
                                                    (short)0, acc10, false, false);
    acc11 = __builtin_amdgcn_wmma_f32_16x16x32_bf16(false, a1, false, b1,
                                                    (short)0, acc11, false, false);
    __syncthreads();
  }

  // --- epilogue: + attn @ bias_p, store f32 ---
  const float* at = attn + b * NF_;
  const int ncol = lane & 15;
  const int mofs = (lane >= 16) ? 8 : 0;
  const int n0 = no + ncol;        // pixel index 0..127
  const int n1 = no + 16 + ncol;
  const int ya = y0 + (n0 >> 6), xa = n0 & 63;
  const int yb = y0 + (n1 >> 6), xb2 = n1 & 63;
#pragma unroll
  for (int r = 0; r < 8; ++r) {
    const int M0 = mo + mofs + r;        // acc0* rows
    const int M1 = mo + 16 + mofs + r;   // acc1* rows
    float bb0 = 0.0f, bb1 = 0.0f;
#pragma unroll
    for (int f = 0; f < NF_; ++f) {
      bb0 = fmaf(at[f], bias_p[f * COUT_ + M0], bb0);
      bb1 = fmaf(at[f], bias_p[f * COUT_ + M1], bb1);
    }
    const long base0 = ((long)b * COUT_ + M0) * (HW_ * HW_);
    const long base1 = ((long)b * COUT_ + M1) * (HW_ * HW_);
    out[base0 + ya * HW_ + xa]  = acc00[r] + bb0;
    out[base0 + yb * HW_ + xb2] = acc01[r] + bb0;
    out[base1 + ya * HW_ + xa]  = acc10[r] + bb1;
    out[base1 + yb * HW_ + xb2] = acc11[r] + bb1;
  }
}

extern "C" void kernel_launch(void* const* d_in, const int* in_sizes, int n_in,
                              void* d_out, int out_size, void* d_ws, size_t ws_size,
                              hipStream_t stream) {
  const float* fci    = (const float*)d_in[0];  // [32,512]
  const float* xin    = (const float*)d_in[1];  // [32,64,64,64]
  const float* fc1_w  = (const float*)d_in[2];  // [256,512]
  const float* fc1_b  = (const float*)d_in[3];  // [256]
  const float* fc2_w  = (const float*)d_in[4];  // [8,256]
  const float* fc2_b  = (const float*)d_in[5];  // [8]
  const float* weight = (const float*)d_in[6];  // [8,64,64,3,3]
  const float* bias_p = (const float*)d_in[7];  // [8,64]

  float* out      = (float*)d_out;                                 // [32,64,64,64]
  float* attn_out = (float*)d_out + (long)B_ * COUT_ * HW_ * HW_;  // [32,8]

  float*  attn_ws = (float*)d_ws;                  // 256 floats
  __bf16* aggw    = (__bf16*)((char*)d_ws + 1024); // [32][64][576] bf16

  routing_kernel<<<B_, 256, 0, stream>>>(fci, fc1_w, fc1_b, fc2_w, fc2_b,
                                         attn_ws, attn_out);

  const long agg_elems = (long)B_ * WPS_;          // 1,179,648
  agg_kernel<<<(int)((agg_elems + 255) / 256), 256, 0, stream>>>(attn_ws, weight, aggw);

  conv_wmma_kernel<<<B_ * (HW_ / 2) * HW_ / HW_, 256, 0, stream>>>(xin, aggw, attn_ws,
                                                                   bias_p, out);
}
